// KPConv_3487513444656
// MI455X (gfx1250) — matrix operand order repaired and checked by reading, not compiled, not run on hardware
//
#include <hip/hip_runtime.h>

typedef __attribute__((ext_vector_type(16))) __bf16 v16bf;
typedef __attribute__((ext_vector_type(8)))  float  v8f;

#define K_PTS   15
#define IN_DIM  32
#define OUT_DIM 32
#define KP_EXTENT 1.2f
#define WAVES_PER_BLOCK 8
#define TILES_PER_WAVE  4

__global__ __launch_bounds__(256) void kpconv_zero(float* __restrict__ out, int n) {
    int i = blockIdx.x * blockDim.x + threadIdx.x;
    if (i < n) out[i] = 0.0f;
}

// Each wave32 processes TILES_PER_WAVE consecutive 16-edge tiles.
// Per tile: D[16x32] += A[16x480] * B[480x32], where the kc-th 32-wide K-chunk
// of A is h[e,kc] * feat[src_e]  (IN_DIM == WMMA bf16 K-chunk == 32).
__global__ __launch_bounds__(256) void kpconv_wmma(
    const float* __restrict__ pos,    // [N,3]
    const float* __restrict__ feat,   // [N,32]
    const int*   __restrict__ src,    // [E]
    const int*   __restrict__ dst,    // [E]
    const float* __restrict__ kp,     // [15,3]
    const float* __restrict__ W,      // [15,32,32]
    float*       __restrict__ out,    // [N,32]
    int n_edges)
{
    // B fragments in LDS, fragment-ordered bf16: [kc][nt][lane][16 elems]
    __shared__ v16bf Wfrag[K_PTS * 2 * 32];      // 30720 bytes
    __shared__ float kplds[K_PTS * 3];

    const int tid = threadIdx.x;
    __bf16* Wb = (__bf16*)Wfrag;

    // ---- cooperative weight pack: wide loads, compile-time trip count ----
    // W has 15*32*32 = 15360 floats = 3840 float4 groups; 256 threads x 15 iters.
    #pragma unroll
    for (int it = 0; it < 15; ++it) {
        const int q    = tid + it * 256;     // float4 index into W
        const int fidx = q * 4;              // flat float index
        const int kc   = fidx >> 10;         // / (32*32)
        const int r    = fidx & 1023;
        const int row  = r >> 5;             // IN index
        const int col0 = r & 31;             // multiple of 4
        const float4 w = *(const float4*)(W + fidx);
        const float wv[4] = { w.x, w.y, w.z, w.w };
        #pragma unroll
        for (int c = 0; c < 4; ++c) {
            const int col  = col0 + c;
            const int nt   = col >> 4;
            const int lane = ((row >> 3) & 1) * 16 + (col & 15);
            const int j    = (row & 7) + ((row >> 4) & 1) * 8;
            Wb[((kc * 2 + nt) * 32 + lane) * 16 + j] = (__bf16)wv[c];
        }
    }
    if (tid < K_PTS * 3) kplds[tid] = kp[tid];
    __syncthreads();

    const int lane = tid & 31;
    const int wave = tid >> 5;
    const int row  = lane & 15;      // edge row within tile (both halves mirror)
    const int hi   = lane >> 4;
    const int col  = lane & 15;
    const float inv_ext = 1.0f / KP_EXTENT;

    const long tileBase = ((long)blockIdx.x * WAVES_PER_BLOCK + wave) * TILES_PER_WAVE;

    #pragma unroll 1
    for (int t = 0; t < TILES_PER_WAVE; ++t) {
        const long eBase = (tileBase + t) * 16;
        if (eBase >= n_edges) break;         // wave-uniform

        // ---- per-edge geometry (row = lane&15; both halves identical) ----
        long e = eBase + row;
        float valid = 1.0f;
        if (e >= n_edges) { e = n_edges - 1; valid = 0.0f; }
        const int s = src[e];
        const int d = dst[e];
        const float yx = pos[(size_t)s * 3 + 0] - pos[(size_t)d * 3 + 0];
        const float yy = pos[(size_t)s * 3 + 1] - pos[(size_t)d * 3 + 1];
        const float yz = pos[(size_t)s * 3 + 2] - pos[(size_t)d * 3 + 2];

        // ---- feat gather in A-fragment layout, converted to bf16 once ----
        const float4* fp4 = (const float4*)(feat + (size_t)s * IN_DIM);
        const float4 f0 = fp4[hi * 2 + 0];
        const float4 f1 = fp4[hi * 2 + 1];
        const float4 f2 = fp4[4 + hi * 2 + 0];
        const float4 f3 = fp4[4 + hi * 2 + 1];
        const float fv[16] = { f0.x, f0.y, f0.z, f0.w, f1.x, f1.y, f1.z, f1.w,
                               f2.x, f2.y, f2.z, f2.w, f3.x, f3.y, f3.z, f3.w };
        v16bf fb;
        #pragma unroll
        for (int j = 0; j < 16; ++j) fb[j] = (__bf16)fv[j];

        v8f c0 = {};   // cols 0..15
        v8f c1 = {};   // cols 16..31

        #pragma unroll
        for (int kc = 0; kc < K_PTS; ++kc) {
            const float dx = yx - kplds[kc * 3 + 0];
            const float dy = yy - kplds[kc * 3 + 1];
            const float dz = yz - kplds[kc * 3 + 2];
            float h = 1.0f - sqrtf(dx * dx + dy * dy + dz * dz) * inv_ext;
            h = (h > 0.0f) ? h : 0.0f;
            h *= valid;

            // A fragment = h * feat, packed bf16 multiply (v_pk_mul_bf16)
            const __bf16 hb = (__bf16)h;
            v16bf hs;
            #pragma unroll
            for (int j = 0; j < 16; ++j) hs[j] = hb;
            const v16bf a = fb * hs;

            const v16bf b0 = Wfrag[(kc * 2 + 0) * 32 + lane];
            const v16bf b1 = Wfrag[(kc * 2 + 1) * 32 + lane];

            c0 = __builtin_amdgcn_wmma_f32_16x16x32_bf16(false, a, false, b0,
                                                         (short)0, c0, false, false);
            c1 = __builtin_amdgcn_wmma_f32_16x16x32_bf16(false, a, false, b1,
                                                         (short)0, c1, false, false);
        }

        // ---- scatter: D VGPR r holds row (r + hi*8), col (lane&15) ----
        #pragma unroll
        for (int r = 0; r < 8; ++r) {
            const int m  = r + hi * 8;
            const int dn = __shfl(d, m);     // dst of edge eBase+m (lanes 0..15 hold rows)
            atomicAdd(&out[(size_t)dn * OUT_DIM + col],      c0[r]);
            atomicAdd(&out[(size_t)dn * OUT_DIM + 16 + col], c1[r]);
        }
    }
}

extern "C" void kernel_launch(void* const* d_in, const int* in_sizes, int n_in,
                              void* d_out, int out_size, void* d_ws, size_t ws_size,
                              hipStream_t stream) {
    const float* pos  = (const float*)d_in[0];
    const float* feat = (const float*)d_in[1];
    const int*   src  = (const int*)d_in[2];
    const int*   dst  = (const int*)d_in[3];
    const float* kp   = (const float*)d_in[4];
    const float* W    = (const float*)d_in[5];
    float* out = (float*)d_out;
    const int n_edges = in_sizes[2];

    kpconv_zero<<<(out_size + 255) / 256, 256, 0, stream>>>(out, out_size);

    const int nTiles  = (n_edges + 15) / 16;
    const int tilesPerBlock = WAVES_PER_BLOCK * TILES_PER_WAVE;
    const int nBlocks = (nTiles + tilesPerBlock - 1) / tilesPerBlock;
    kpconv_wmma<<<nBlocks, 256, 0, stream>>>(pos, feat, src, dst, kp, W, out, n_edges);
}
